// SimpleGCN_43971875177084
// MI455X (gfx1250) — matrix operand order repaired and checked by reading, not compiled, not run on hardware
//
#include <hip/hip_runtime.h>
#include <hip/hip_bf16.h>

// ---------------------------------------------------------------------------
// 2-layer GCN for gfx1250 (MI455X).
//   h1  = X @ W1                       (bf16 WMMA, f32 accum)
//   agg1= scatter_add(norm * h1[src])  (global_atomic_add_f32, L2-resident)
//   H   = relu(agg1 + d_i^2*h1 + b1)   (fused, emits bf16 for next GEMM)
//   h2  = H @ W2                       (bf16 WMMA)
//   out = scatter_add(...) + d_i^2*h2 + b2
// ---------------------------------------------------------------------------

#define NODES_FDIM1 128
#define NODES_FDIM2 64

typedef __attribute__((ext_vector_type(16))) __bf16 v16bf;
typedef __attribute__((ext_vector_type(8)))  float  v8f;

union Frag16 { v16bf v; unsigned short u[16]; };

__device__ __forceinline__ unsigned short f2bf(float f) {
  unsigned int u = __float_as_uint(f);
  unsigned int r = u + 0x7FFFu + ((u >> 16) & 1u);   // round-to-nearest-even
  return (unsigned short)(r >> 16);
}

// ---- init: deg=1 (self loop), zero agg1 and out --------------------------
__global__ void gcn_init(float* __restrict__ deg, float* __restrict__ agg1,
                         float* __restrict__ out, int N) {
  long long total = (long long)N * NODES_FDIM1;
  for (long long t = (long long)blockIdx.x * blockDim.x + threadIdx.x;
       t < total; t += (long long)gridDim.x * blockDim.x) {
    agg1[t] = 0.0f;
    if (t < (long long)N * NODES_FDIM2) out[t] = 0.0f;
    if (t < N) deg[t] = 1.0f;
  }
}

// ---- degree accumulation over edge destinations --------------------------
__global__ void gcn_deg(const int* __restrict__ dst, float* __restrict__ deg, int E) {
  for (int e = blockIdx.x * blockDim.x + threadIdx.x; e < E;
       e += gridDim.x * blockDim.x)
    atomicAdd(&deg[dst[e]], 1.0f);
}

__global__ void gcn_deg_rsqrt(float* __restrict__ deg, int N) {
  int t = blockIdx.x * blockDim.x + threadIdx.x;
  if (t < N) deg[t] = rsqrtf(deg[t]);   // deg >= 1 always (self loop)
}

// ---- f32 -> bf16 conversion of X, W1, W2 ---------------------------------
__global__ void gcn_cvt(const float* __restrict__ x, const float* __restrict__ W1,
                        const float* __restrict__ W2,
                        unsigned short* __restrict__ xbf,
                        unsigned short* __restrict__ w1bf,
                        unsigned short* __restrict__ w2bf, long long nx) {
  for (long long t = (long long)blockIdx.x * blockDim.x + threadIdx.x;
       t < nx; t += (long long)gridDim.x * blockDim.x) {
    xbf[t] = f2bf(x[t]);
    if (t < NODES_FDIM1 * NODES_FDIM1) w1bf[t] = f2bf(W1[t]);
    if (t < NODES_FDIM1 * NODES_FDIM2) w2bf[t] = f2bf(W2[t]);
  }
}

// ---- bf16 WMMA GEMM: C[M,N] = A[M,K] @ B[K,N] ----------------------------
// blockIdx.x = 16-row M tile; each wave owns one 16-col N tile.
// A fragment layout per ISA 7.12.2 (16-bit A 16x32); B fragment: lane = N column,
// K order per element matching A (K 0..7,16..23 or 8..15,24..31).
__global__ void gcn_gemm_bf16(const unsigned short* __restrict__ A,
                              const unsigned short* __restrict__ B,
                              float* __restrict__ C, int M, int K, int N) {
  const int lane = threadIdx.x & 31;
  const int wave = threadIdx.x >> 5;
  const int tm   = blockIdx.x << 4;
  const int tn   = wave << 4;
  if (tm >= M || tn >= N) return;

  const int arow = tm + (lane & 15);
  const int akb  = (lane < 16) ? 0 : 8;

  v8f acc = {};
#pragma unroll
  for (int k0 = 0; k0 < 128; k0 += 32) {   // K == 128 for both layers
    Frag16 a, b;
    const unsigned short* ap = A + (size_t)arow * K + k0 + akb;
    __builtin_memcpy(&a.u[0], ap,      16);   // K 0..7   (or 8..15)
    __builtin_memcpy(&a.u[8], ap + 16, 16);   // K 16..23 (or 24..31)
    const unsigned short* bp = B + (size_t)(k0 + akb) * N + tn + (lane & 15);
#pragma unroll
    for (int s = 0; s < 16; ++s)
      b.u[s] = bp[(size_t)((s < 8) ? s : (s + 8)) * N];
    acc = __builtin_amdgcn_wmma_f32_16x16x32_bf16(false, a.v, false, b.v,
                                                  (short)0, acc, false, false);
  }

  const int ro  = (lane < 16) ? 0 : 8;
  const int col = tn + (lane & 15);
#pragma unroll
  for (int r = 0; r < 8; ++r)
    C[(size_t)(tm + ro + r) * N + col] = acc[r];
}

// ---- edge scatter, 128 features: one wave per edge, float4 per lane ------
__global__ void gcn_scatter128(const int* __restrict__ src, const int* __restrict__ dst,
                               const float* __restrict__ dis, const float* __restrict__ h,
                               float* __restrict__ agg, int E) {
  const int lane  = threadIdx.x & 31;
  const int warp  = (blockIdx.x * blockDim.x + threadIdx.x) >> 5;
  const int nwarp = (gridDim.x * blockDim.x) >> 5;
  for (int e = warp; e < E; e += nwarp) {
    const int s = src[e], d = dst[e];
    const int en = e + nwarp;
    if (en < E)  // hide gather latency: prefetch next source row (global_prefetch)
      __builtin_prefetch(h + (size_t)src[en] * NODES_FDIM1, 0, 0);
    const float w = dis[s] * dis[d];
    float4 v = reinterpret_cast<const float4*>(h + (size_t)s * NODES_FDIM1)[lane];
    float* o = agg + (size_t)d * NODES_FDIM1 + lane * 4;
    atomicAdd(o + 0, v.x * w);
    atomicAdd(o + 1, v.y * w);
    atomicAdd(o + 2, v.z * w);
    atomicAdd(o + 3, v.w * w);
  }
}

// ---- edge scatter, 64 features: one wave per edge, float2 per lane -------
__global__ void gcn_scatter64(const int* __restrict__ src, const int* __restrict__ dst,
                              const float* __restrict__ dis, const float* __restrict__ h,
                              float* __restrict__ agg, int E) {
  const int lane  = threadIdx.x & 31;
  const int warp  = (blockIdx.x * blockDim.x + threadIdx.x) >> 5;
  const int nwarp = (gridDim.x * blockDim.x) >> 5;
  for (int e = warp; e < E; e += nwarp) {
    const int s = src[e], d = dst[e];
    const int en = e + nwarp;
    if (en < E)
      __builtin_prefetch(h + (size_t)src[en] * NODES_FDIM2, 0, 0);
    const float w = dis[s] * dis[d];
    float2 v = reinterpret_cast<const float2*>(h + (size_t)s * NODES_FDIM2)[lane];
    float* o = agg + (size_t)d * NODES_FDIM2 + lane * 2;
    atomicAdd(o + 0, v.x * w);
    atomicAdd(o + 1, v.y * w);
  }
}

// ---- layer-1 finalize: self loop + bias + relu, emit bf16 ----------------
__global__ void gcn_fin1(const float* __restrict__ agg, const float* __restrict__ h1,
                         const float* __restrict__ dis, const float* __restrict__ b1,
                         unsigned short* __restrict__ hbf, int N) {
  long long total = (long long)N * NODES_FDIM1;
  for (long long t = (long long)blockIdx.x * blockDim.x + threadIdx.x;
       t < total; t += (long long)gridDim.x * blockDim.x) {
    const int i = (int)(t >> 7);
    const int f = (int)(t & 127);
    const float di = dis[i];
    float v = agg[t] + h1[t] * di * di + b1[f];
    hbf[t] = f2bf(fmaxf(v, 0.0f));
  }
}

// ---- layer-2 finalize: self loop + bias into d_out -----------------------
__global__ void gcn_fin2(float* __restrict__ out, const float* __restrict__ h2,
                         const float* __restrict__ dis, const float* __restrict__ b2,
                         int N) {
  long long total = (long long)N * NODES_FDIM2;
  for (long long t = (long long)blockIdx.x * blockDim.x + threadIdx.x;
       t < total; t += (long long)gridDim.x * blockDim.x) {
    const int i = (int)(t >> 6);
    const int f = (int)(t & 63);
    const float di = dis[i];
    out[t] = out[t] + h2[t] * di * di + b2[f];
  }
}

extern "C" void kernel_launch(void* const* d_in, const int* in_sizes, int n_in,
                              void* d_out, int out_size, void* d_ws, size_t ws_size,
                              hipStream_t stream) {
  const float* x  = (const float*)d_in[0];
  const int*   ei = (const int*)d_in[1];      // [2, E] row-major: src then dst
  const float* W1 = (const float*)d_in[2];
  const float* b1 = (const float*)d_in[3];
  const float* W2 = (const float*)d_in[4];
  const float* b2 = (const float*)d_in[5];

  const int N = in_sizes[0] / NODES_FDIM1;    // 50000
  const int E = in_sizes[1] / 2;              // 800000
  const int* src = ei;
  const int* dst = ei + E;
  float* out = (float*)d_out;

  // ---- workspace carve-up (all offsets 256B aligned) ----
  char* p = (char*)d_ws;
  auto carve = [&](size_t bytes) -> void* {
    void* r = (void*)p;
    p += (bytes + 255) & ~(size_t)255;
    return r;
  };
  float*          deg   = (float*)         carve((size_t)N * 4);
  unsigned short* xbf   = (unsigned short*)carve((size_t)N * NODES_FDIM1 * 2);
  unsigned short* w1bf  = (unsigned short*)carve(NODES_FDIM1 * NODES_FDIM1 * 2);
  unsigned short* w2bf  = (unsigned short*)carve(NODES_FDIM1 * NODES_FDIM2 * 2);
  float*          h1    = (float*)         carve((size_t)N * NODES_FDIM1 * 4);
  float*          agg1  = (float*)         carve((size_t)N * NODES_FDIM1 * 4);
  float*          h2    = (float*)         carve((size_t)N * NODES_FDIM2 * 4);
  unsigned short* hbf   = xbf;  // X (bf16) is dead after GEMM1; reuse for H

  const long long nx = (long long)N * NODES_FDIM1;
  const int blk = 256;
  const int gInit = (int)((nx + blk - 1) / blk);
  const int gE    = (E + blk - 1) / blk;

  // 1) init accumulators + self-loop degree
  gcn_init<<<gInit, blk, 0, stream>>>(deg, agg1, out, N);
  // 2) degree count, 3) rsqrt
  gcn_deg<<<gE, blk, 0, stream>>>(dst, deg, E);
  gcn_deg_rsqrt<<<(N + blk - 1) / blk, blk, 0, stream>>>(deg, N);
  // 4) bf16 conversion
  gcn_cvt<<<gInit, blk, 0, stream>>>(x, W1, W2, xbf, w1bf, w2bf, nx);
  // 5) GEMM1: h1 = X @ W1   (M tiles = N/16, 8 waves of 16 cols)
  gcn_gemm_bf16<<<N / 16, (NODES_FDIM1 / 16) * 32, 0, stream>>>(
      xbf, w1bf, h1, N, NODES_FDIM1, NODES_FDIM1);
  // 6) edge scatter into agg1 (wave per edge, 8 edges/block)
  gcn_scatter128<<<(E + 7) / 8, blk, 0, stream>>>(src, dst, deg, h1, agg1, E);
  // 7) self loop + bias + relu -> bf16 H
  gcn_fin1<<<gInit, blk, 0, stream>>>(agg1, h1, deg, b1, hbf, N);
  // 8) GEMM2: h2 = H @ W2   (4 waves of 16 cols)
  gcn_gemm_bf16<<<N / 16, (NODES_FDIM2 / 16) * 32, 0, stream>>>(
      hbf, w2bf, h2, N, NODES_FDIM1, NODES_FDIM2);
  // 9) edge scatter into out
  gcn_scatter64<<<(E + 7) / 8, blk, 0, stream>>>(src, dst, deg, h2, out, E);
  // 10) self loop + bias
  gcn_fin2<<<(int)(((long long)N * NODES_FDIM2 + blk - 1) / blk), blk, 0, stream>>>(
      out, h2, deg, b2, N);
}
